// MultiHeadAttention_71777493451213
// MI455X (gfx1250) — compile-verified, matrix operand order checked by reading, not run on hardware
//
#include <hip/hip_runtime.h>

// MI455X / gfx1250 fused multi-head attention, wave32 + WMMA f16.
// B=2 S=2048 C=1024 H=16 D=64.

#define B_ 2
#define S_ 2048
#define C_ 1024
#define H_ 16
#define D_ 64

typedef __attribute__((ext_vector_type(16))) _Float16 v16h;
typedef __attribute__((ext_vector_type(8)))  _Float16 v8h;
typedef __attribute__((ext_vector_type(8)))  float    v8f;

// Load a 16x32-f16 A/B fragment half for this lane per the CDNA5 layout:
// lanes 0-15 hold K=0..7 (vgpr0-3) and K=16..23 (vgpr4-7);
// lanes 16-31 hold K=8..15 and K=24..31.  p points at (row, k=0) of a
// row-major tile; both 8-half groups are contiguous -> two b128 loads.
__device__ __forceinline__ v16h load_frag(const _Float16* p, int hf) {
  v8h lo = *(const v8h*)(p + 8 * hf);
  v8h hi = *(const v8h*)(p + 16 + 8 * hf);
  v16h r;
#pragma unroll
  for (int i = 0; i < 8; ++i) { r[i] = lo[i]; r[i + 8] = hi[i]; }
  return r;
}

__device__ __forceinline__ v8f wmma_f16(v16h a, v16h b, v8f c) {
  return __builtin_amdgcn_wmma_f32_16x16x32_f16(false, a, false, b,
                                                (short)0, c, false, false);
}

// ---------------------------------------------------------------------------
// Kernel 1: fused QKV projection.  out = x @ W^T + bias, downconvert to f16.
// grid.z picks {Q,K,V}.  Q,K written [b,h,s,d]; V written transposed [b,h,d,s]
// so the context GEMM gets contiguous B-fragments.
// 64x64 output tile per workgroup, 4 waves.  K-chunks of 64 with
// double-buffered LDS: chunk c+1's global loads are issued as a clause
// before chunk c's WMMAs, converted to f16 into the alternate buffer after.
// All staging addresses are one base pointer + immediate offsets.
// ---------------------------------------------------------------------------
#define XSTRIDE 72   // 64 cols + 8 pad halfs; 144B rows keep 16B alignment

__global__ __launch_bounds__(128) void qkv_proj_kernel(
    const float* __restrict__ xq, const float* __restrict__ xk,
    const float* __restrict__ xv,
    const float* __restrict__ Wq, const float* __restrict__ bq,
    const float* __restrict__ Wk, const float* __restrict__ bk,
    const float* __restrict__ Wv, const float* __restrict__ bv,
    _Float16* __restrict__ Qh, _Float16* __restrict__ Kh,
    _Float16* __restrict__ VhT) {
  const int z = blockIdx.z;
  const float* x    = (z == 0) ? xq : (z == 1) ? xk : xv;
  const float* W    = (z == 0) ? Wq : (z == 1) ? Wk : Wv;
  const float* bias = (z == 0) ? bq : (z == 1) ? bk : bv;
  _Float16* out     = (z == 0) ? Qh : (z == 1) ? Kh : VhT;

  const int m0   = blockIdx.x * 64;
  const int n0   = blockIdx.y * 64;
  const int tid  = threadIdx.x;
  const int w    = tid >> 5;
  const int lane = tid & 31;
  const int l16  = lane & 15;
  const int hf   = lane >> 4;

  __shared__ _Float16 Xs[2][64 * XSTRIDE];   // 2 x 9 KB
  __shared__ _Float16 Ws[2][64 * XSTRIDE];   // 2 x 9 KB

  // Staging geometry: thread owns rows row0+8j (j=0..7), fixed col group.
  const int row0 = tid >> 4;                 // 0..7
  const int col  = (tid & 15) << 2;          // 0,4,..,60
  const float* xbase = x + (size_t)(m0 + row0) * C_ + col;
  const float* wbase = W + (size_t)(n0 + row0) * C_ + col;

  float4 rx[8], rw[8];
  // Prologue: load chunk 0 and stage into buffer 0.
#pragma unroll
  for (int j = 0; j < 8; ++j) {
    rx[j] = *(const float4*)(xbase + j * 8 * C_);   // immediate offsets
    rw[j] = *(const float4*)(wbase + j * 8 * C_);
  }
  {
    _Float16* xs = &Xs[0][row0 * XSTRIDE + col];
    _Float16* ws = &Ws[0][row0 * XSTRIDE + col];
#pragma unroll
    for (int j = 0; j < 8; ++j) {
      _Float16* d = xs + j * 8 * XSTRIDE;
      d[0] = (_Float16)rx[j].x; d[1] = (_Float16)rx[j].y;
      d[2] = (_Float16)rx[j].z; d[3] = (_Float16)rx[j].w;
      d = ws + j * 8 * XSTRIDE;
      d[0] = (_Float16)rw[j].x; d[1] = (_Float16)rw[j].y;
      d[2] = (_Float16)rw[j].z; d[3] = (_Float16)rw[j].w;
    }
  }

  v8f acc[4] = {};

  for (int c = 0; c < 16; ++c) {           // 16 chunks of 64 over C_=1024
    const int cur = c & 1;
    __syncthreads();                        // chunk c staged & prior reads done

    // Issue chunk c+1 global loads up front (overlap with WMMA below).
    if (c < 15) {
      const int k1 = (c + 1) * 64;
#pragma unroll
      for (int j = 0; j < 8; ++j) {
        rx[j] = *(const float4*)(xbase + k1 + j * 8 * C_);
        rw[j] = *(const float4*)(wbase + k1 + j * 8 * C_);
      }
    }

    // Compute on chunk c from buf[cur].
    const _Float16* xrow = &Xs[cur][(w * 16 + l16) * XSTRIDE];
    v16h a0 = load_frag(xrow, hf);
    v16h a1 = load_frag(xrow + 32, hf);
    const _Float16* wrow = &Ws[cur][l16 * XSTRIDE];
#pragma unroll
    for (int nt = 0; nt < 4; ++nt) {
      v16h b0 = load_frag(wrow + nt * 16 * XSTRIDE, hf);
      v16h b1 = load_frag(wrow + nt * 16 * XSTRIDE + 32, hf);
      acc[nt] = wmma_f16(a0, b0, acc[nt]);
      acc[nt] = wmma_f16(a1, b1, acc[nt]);
    }

    // Convert + stage chunk c+1 into the alternate buffer.
    if (c < 15) {
      const int nxt = cur ^ 1;
      _Float16* xs = &Xs[nxt][row0 * XSTRIDE + col];
      _Float16* ws = &Ws[nxt][row0 * XSTRIDE + col];
#pragma unroll
      for (int j = 0; j < 8; ++j) {
        _Float16* d = xs + j * 8 * XSTRIDE;
        d[0] = (_Float16)rx[j].x; d[1] = (_Float16)rx[j].y;
        d[2] = (_Float16)rx[j].z; d[3] = (_Float16)rx[j].w;
        d = ws + j * 8 * XSTRIDE;
        d[0] = (_Float16)rw[j].x; d[1] = (_Float16)rw[j].y;
        d[2] = (_Float16)rw[j].z; d[3] = (_Float16)rw[j].w;
      }
    }
  }

#pragma unroll
  for (int nt = 0; nt < 4; ++nt) {
    int n = n0 + nt * 16 + l16;
    int h = n >> 6;
    int d = n & 63;
    float bias_n = bias[n];
#pragma unroll
    for (int r = 0; r < 8; ++r) {
      int m = m0 + w * 16 + r + 8 * hf;   // C/D layout: vgpr r -> M=r (+8 hi)
      int b = m >> 11;
      int s = m & (S_ - 1);
      float val = acc[nt][r] + bias_n;
      size_t o;
      if (z < 2) o = (((size_t)(b * H_ + h) * S_ + s) << 6) + d;   // [bh][s][d]
      else       o = (((size_t)(b * H_ + h) * D_ + d) << 11) + s;  // [bh][d][s]
      out[o] = (_Float16)val;             // re-read by kernel 2: keep RT
    }
  }
}

// ---------------------------------------------------------------------------
// Kernel 2: fused scores + softmax + attn write + context GEMM.
// One workgroup (8 waves) per (b, h, 16-row q tile).  Wave w owns the
// 256-key strip [w*256, w*256+256): 16 WMMA score tiles held in registers.
// NOTE: both WMMA loops must be FULLY unrolled so the accumulator arrays
// stay register-resident (partial unroll demotes them to scratch).
// ---------------------------------------------------------------------------
__global__ __launch_bounds__(256) void attn_fused_kernel(
    const _Float16* __restrict__ Qh, const _Float16* __restrict__ Kh,
    const _Float16* __restrict__ VhT,
    float* __restrict__ ctx, float* __restrict__ attn) {
  const int qt = blockIdx.x & 127;
  const int bh = blockIdx.x >> 7;           // b*H + h
  const int b  = bh >> 4;
  const int h  = bh & 15;
  const int s0 = qt * 16;

  const int tid  = threadIdx.x;
  const int w    = tid >> 5;
  const int lane = tid & 31;
  const int l16  = lane & 15;
  const int hf   = lane >> 4;

  const _Float16* Qb = Qh  + (size_t)bh * S_ * D_;
  const _Float16* Kb = Kh  + (size_t)bh * S_ * D_;
  const _Float16* Vt = VhT + (size_t)bh * D_ * S_;

  __shared__ _Float16 pf[16 * S_];          // 64 KB f16 probability tile
  __shared__ float red[8][16];
  __shared__ float gstat[16];
  __shared__ float cred[4][8 * 32];

  // Long-range prefetch: this wave's V strip is needed only after softmax.
  const int tile  = w & 3;                  // which 16-wide d slice (phase D)
  const int khalf = w >> 2;                 // which 1024-key half  (phase D)
  const _Float16* vrow = Vt + (size_t)(tile * 16 + l16) * S_ + khalf * 1024;
  __builtin_prefetch(vrow, 0, 0);

  // --- scores: 16 x 2048, this wave's 16x256 strip in registers ----------
  const _Float16* qrow = Qb + (size_t)(s0 + l16) * D_;
  v16h a0 = load_frag(qrow, hf);
  v16h a1 = load_frag(qrow + 32, hf);

  const _Float16* kstrip = Kb + (size_t)(w * 256 + l16) * D_;  // + t*1024
  v8f acc[16] = {};
#pragma unroll
  for (int t = 0; t < 16; ++t) {
    v16h b0 = load_frag(kstrip + t * 16 * D_, hf);      // immediate offsets
    v16h b1 = load_frag(kstrip + t * 16 * D_ + 32, hf);
    acc[t] = wmma_f16(a0, b0, acc[t]);
    acc[t] = wmma_f16(a1, b1, acc[t]);
  }

  // --- softmax: scale, row max, exp, row sum ------------------------------
  const float scale = 0.125f;               // 1/sqrt(64)
  float rmax[8];
#pragma unroll
  for (int r = 0; r < 8; ++r) rmax[r] = -3.402823466e+38f;
#pragma unroll
  for (int t = 0; t < 16; ++t)
#pragma unroll
    for (int r = 0; r < 8; ++r) {
      acc[t][r] *= scale;
      rmax[r] = fmaxf(rmax[r], acc[t][r]);
    }
#pragma unroll
  for (int m = 1; m < 16; m <<= 1)
#pragma unroll
    for (int r = 0; r < 8; ++r)
      rmax[r] = fmaxf(rmax[r], __shfl_xor(rmax[r], m, 32));
  if (l16 == 0)
#pragma unroll
    for (int r = 0; r < 8; ++r) red[w][hf * 8 + r] = rmax[r];
  __syncthreads();
  if (tid < 16) {
    float g = -3.402823466e+38f;
#pragma unroll
    for (int ww = 0; ww < 8; ++ww) g = fmaxf(g, red[ww][tid]);
    gstat[tid] = g;
  }
  __syncthreads();
  float gm[8];
#pragma unroll
  for (int r = 0; r < 8; ++r) gm[r] = gstat[hf * 8 + r];

  float rsum[8] = {0.f, 0.f, 0.f, 0.f, 0.f, 0.f, 0.f, 0.f};
#pragma unroll
  for (int t = 0; t < 16; ++t)
#pragma unroll
    for (int r = 0; r < 8; ++r) {
      float p = __expf(acc[t][r] - gm[r]);   // v_exp_f32, co-executes w/ WMMA
      acc[t][r] = p;
      rsum[r] += p;
    }
#pragma unroll
  for (int m = 1; m < 16; m <<= 1)
#pragma unroll
    for (int r = 0; r < 8; ++r)
      rsum[r] += __shfl_xor(rsum[r], m, 32);
  if (l16 == 0)
#pragma unroll
    for (int r = 0; r < 8; ++r) red[w][hf * 8 + r] = rsum[r];
  __syncthreads();
  if (tid < 16) {
    float g = 0.f;
#pragma unroll
    for (int ww = 0; ww < 8; ++ww) g += red[ww][tid];
    gstat[tid] = 1.0f / g;
  }
  __syncthreads();
  float rinv[8];
#pragma unroll
  for (int r = 0; r < 8; ++r) rinv[r] = gstat[hf * 8 + r];

  // --- single attn write (fp32, non-temporal) + pack f16 probs into LDS ---
  float* arow = attn + ((size_t)(h * B_ + b) * S_ + s0) * S_ + w * 256 + l16;
  _Float16* prow_o = &pf[w * 256 + l16];
#pragma unroll
  for (int t = 0; t < 16; ++t) {
#pragma unroll
    for (int r = 0; r < 8; ++r) {
      int row = hf * 8 + r;
      float p = acc[t][r] * rinv[r];
      __builtin_nontemporal_store(p, arow + (size_t)row * S_ + t * 16);
      prow_o[row * S_ + t * 16] = (_Float16)p;
    }
  }
  __syncthreads();

  // --- context = P @ V (contraction over 2048 keys, split 2 ways) ---------
  const int kbase = khalf * 1024;
  v8f cacc = {};
  const _Float16* prow = &pf[l16 * S_ + kbase];
#pragma unroll
  for (int kc = 0; kc < 1024; kc += 32) {
    v16h a  = load_frag(prow + kc, hf);
    v16h bf = load_frag(vrow + kc, hf);
    cacc = wmma_f16(a, bf, cacc);
  }
  if (khalf == 1)
#pragma unroll
    for (int r = 0; r < 8; ++r) cred[tile][r * 32 + lane] = cacc[r];
  __syncthreads();
  if (khalf == 0) {
#pragma unroll
    for (int r = 0; r < 8; ++r) {
      float val = cacc[r] + cred[tile][r * 32 + lane];
      int row = s0 + r + 8 * hf;
      int col = h * 64 + tile * 16 + l16;
      __builtin_nontemporal_store(val, &ctx[((size_t)b * S_ + row) * C_ + col]);
    }
  }
}

// ---------------------------------------------------------------------------
extern "C" void kernel_launch(void* const* d_in, const int* in_sizes, int n_in,
                              void* d_out, int out_size, void* d_ws,
                              size_t ws_size, hipStream_t stream) {
  const float* q  = (const float*)d_in[0];
  const float* k  = (const float*)d_in[1];
  const float* v  = (const float*)d_in[2];
  const float* Wq = (const float*)d_in[3];
  const float* bq = (const float*)d_in[4];
  const float* Wk = (const float*)d_in[5];
  const float* bk = (const float*)d_in[6];
  const float* Wv = (const float*)d_in[7];
  const float* bv = (const float*)d_in[8];

  float* ctx  = (float*)d_out;                       // [B,S,C]
  float* attn = ctx + (size_t)B_ * S_ * C_;          // [H*B,S,S]

  const size_t N = (size_t)B_ * H_ * S_ * D_;        // 4,194,304 halfs each
  _Float16* Qh  = (_Float16*)d_ws;                   // [b,h,s,d]
  _Float16* Kh  = Qh + N;                            // [b,h,s,d]
  _Float16* VhT = Kh + N;                            // [b,h,d,s]

  qkv_proj_kernel<<<dim3(64, 16, 3), 128, 0, stream>>>(
      q, k, v, Wq, bq, Wk, bk, Wv, bv, Qh, Kh, VhT);

  attn_fused_kernel<<<dim3(B_ * H_ * (S_ / 16)), 256, 0, stream>>>(
      Qh, Kh, VhT, ctx, attn);
}
